// SelfAttention_61778809586166
// MI455X (gfx1250) — compile-verified, matrix operand order checked by reading, not run on hardware
//
#include <hip/hip_runtime.h>
#include <hip/hip_bf16.h>

// ---------------------------------------------------------------------------
// Self-attention for B=4, S=2048, D=1024 on gfx1250 (MI455X).
// All matrix math via v_wmma_f32_16x16x32_f16 (f32 accumulate).
//   1) cvt_f16      : x fp32 -> f16
//   2) transpose_w  : W fp32 [K,N] -> Wt f16 [N,K]   (x3)
//   3) gemm16       : Kh = x@Wk, Qh = x@Wq (row-major f16),
//                     Vt = (x@Wv)^T per batch (f16, [D,S]); 32x64 wave tiles
//   4) attn_scores  : per 16-row strip: WMMA K.Q^T/sqrt(D) into 128KB LDS
//                     (k-outer, 16 accumulators, A loaded once per k-step),
//                     fused row softmax, write P f16
//   5) attn_out     : out = P @ V (A=P, Bt=Vt), 32x64 wave tiles, f32 out
// All GEMMs use A row-major [M,K] + B^T row-major [N,K] so every fragment is
// two contiguous 16B loads per lane, matching the CDNA5 16-bit A/B layout.
// ---------------------------------------------------------------------------

typedef _Float16 half_t;
typedef __attribute__((ext_vector_type(16))) _Float16 v16h;
typedef __attribute__((ext_vector_type(8)))  float    v8f;

#define WMMA_F16(a, b, c) \
  __builtin_amdgcn_wmma_f32_16x16x32_f16(false, (a), false, (b), (short)0, (c), false, false)

static constexpr int BB = 4;
static constexpr int SS = 2048;
static constexpr int DD = 1024;

// Load a 16x32 f16 fragment (A-style) for one lane.
// CDNA5 layout: lane = r + 16*h (r = M or N index, h = K-half select).
//   elems 0..7  = K[8h .. 8h+7], elems 8..15 = K[16+8h .. 16+8h+7]
__device__ __forceinline__ v16h load_frag(const half_t* __restrict__ base,
                                          int stride, int row, int k0, int h) {
  const half_t* p = base + (size_t)row * (size_t)stride + (size_t)(k0 + 8 * h);
  union { v16h v; uint4 u[2]; } f;
  f.u[0] = *(const uint4*)(p);
  f.u[1] = *(const uint4*)(p + 16);
  return f.v;
}

// ---------------------------------------------------------------------------
// 1) fp32 -> f16 elementwise (vectorized x4)
// ---------------------------------------------------------------------------
__global__ __launch_bounds__(256) void cvt_f16(const float* __restrict__ in,
                                               half_t* __restrict__ out, int n4) {
  int i = blockIdx.x * blockDim.x + threadIdx.x;
  if (i >= n4) return;
  float4 v = ((const float4*)in)[i];
  union { half_t h[4]; uint2 u; } o;
  o.h[0] = (half_t)v.x; o.h[1] = (half_t)v.y;
  o.h[2] = (half_t)v.z; o.h[3] = (half_t)v.w;
  ((uint2*)out)[i] = o.u;
}

// ---------------------------------------------------------------------------
// 2) W [K=1024, N=1024] fp32 -> Wt [N, K] f16 (LDS tile transpose)
// ---------------------------------------------------------------------------
__global__ __launch_bounds__(256) void transpose_w(const float* __restrict__ W,
                                                   half_t* __restrict__ Wt) {
  __shared__ float tile[32][33];
  int kb = blockIdx.x * 32, nb = blockIdx.y * 32;
  int tx = threadIdx.x & 31, ty = threadIdx.x >> 5;   // 8 rows of 32
#pragma unroll
  for (int yy = ty; yy < 32; yy += 8)
    tile[yy][tx] = W[(size_t)(kb + yy) * DD + nb + tx];
  __syncthreads();
#pragma unroll
  for (int yy = ty; yy < 32; yy += 8)
    Wt[(size_t)(nb + yy) * DD + kb + tx] = (half_t)tile[tx][yy];
}

// ---------------------------------------------------------------------------
// 3) GEMM: C[M,N] = A[M,K] * Bt[N,K]^T, f16 in, f16 out.
//    Wave tile = 32x64: 2 A-frags + 4 B-frags -> 8 WMMAs per 32-wide k-step
//    (12 b128 loads per 8 WMMAs).
//    vmode 0: C row-major [M,N]
//    vmode 1: C transposed per batch: Ct[b][N, S] (for V^T), S = SS
// ---------------------------------------------------------------------------
__global__ __launch_bounds__(256) void gemm16(const half_t* __restrict__ A,
                                              const half_t* __restrict__ Bt,
                                              half_t* __restrict__ C,
                                              int M, int N, int K, int vmode) {
  const int lane = threadIdx.x & 31;
  const int wave = threadIdx.x >> 5;
  const int gw   = blockIdx.x * (blockDim.x >> 5) + wave;
  const int wavesN = N >> 6;                  // 64 cols per wave
  const int mt = gw / wavesN;
  const int nt = gw % wavesN;
  const int m0 = mt * 32;
  const int n0 = nt * 64;
  if (m0 >= M) return;
  const int r = lane & 15, h = lane >> 4;

  v8f acc[4][2];
#pragma unroll
  for (int j = 0; j < 4; ++j) {
    acc[j][0] = (v8f)(0.0f);
    acc[j][1] = (v8f)(0.0f);
  }

  for (int k0 = 0; k0 < K; k0 += 32) {
    v16h a0 = load_frag(A,  K, m0 + r,      k0, h);
    v16h a1 = load_frag(A,  K, m0 + 16 + r, k0, h);
    v16h b0 = load_frag(Bt, K, n0 + r,      k0, h);
    v16h b1 = load_frag(Bt, K, n0 + 16 + r, k0, h);
    v16h b2 = load_frag(Bt, K, n0 + 32 + r, k0, h);
    v16h b3 = load_frag(Bt, K, n0 + 48 + r, k0, h);
    acc[0][0] = WMMA_F16(a0, b0, acc[0][0]);
    acc[0][1] = WMMA_F16(a1, b0, acc[0][1]);
    acc[1][0] = WMMA_F16(a0, b1, acc[1][0]);
    acc[1][1] = WMMA_F16(a1, b1, acc[1][1]);
    acc[2][0] = WMMA_F16(a0, b2, acc[2][0]);
    acc[2][1] = WMMA_F16(a1, b2, acc[2][1]);
    acc[3][0] = WMMA_F16(a0, b3, acc[3][0]);
    acc[3][1] = WMMA_F16(a1, b3, acc[3][1]);
  }

  if (vmode == 0) {
    // C fragment: lane holds N = r, rows M = 16*mi + 8h + g
#pragma unroll
    for (int j = 0; j < 4; ++j) {
#pragma unroll
      for (int mi = 0; mi < 2; ++mi) {
#pragma unroll
        for (int g = 0; g < 8; ++g) {
          int m = m0 + 16 * mi + 8 * h + g;
          C[(size_t)m * N + n0 + 16 * j + r] = (half_t)acc[j][mi][g];
        }
      }
    }
  } else {
    // transposed store: row of Ct = output feature n; 8 contiguous halfs/lane
    int b    = m0 / SS;
    int mloc = m0 - b * SS;
    half_t* Cb = C + (size_t)b * N * SS;
#pragma unroll
    for (int j = 0; j < 4; ++j) {
      int row = n0 + 16 * j + r;
#pragma unroll
      for (int mi = 0; mi < 2; ++mi) {
        union { half_t hh[8]; uint4 u; } pk;
#pragma unroll
        for (int g = 0; g < 8; ++g) pk.hh[g] = (half_t)acc[j][mi][g];
        *(uint4*)(Cb + (size_t)row * SS + mloc + 16 * mi + 8 * h) = pk.u;
      }
    }
  }
}

// ---------------------------------------------------------------------------
// 4) Fused scores + softmax.
//    Block = (b, 16-row strip). 8 waves each own 256 score columns.
//    k-outer loop: A-fragment loaded ONCE per k-step, 16 live accumulators.
//    Scores strip lives in 16 x 2048 f32 dynamic LDS (128 KB; WGP has 320KB).
// ---------------------------------------------------------------------------
__global__ __launch_bounds__(256) void attn_scores(const half_t* __restrict__ Kh,
                                                   const half_t* __restrict__ Qh,
                                                   half_t* __restrict__ Ph,
                                                   float scale) {
  extern __shared__ float smem[];               // [16][2048]
  const int b  = blockIdx.y;
  const int m0 = blockIdx.x * 16;
  const half_t* Kb = Kh + (size_t)b * SS * DD;
  const half_t* Qb = Qh + (size_t)b * SS * DD;
  const int lane = threadIdx.x & 31, wave = threadIdx.x >> 5;
  const int r = lane & 15, h = lane >> 4;
  const int t0 = wave * 256;                    // this wave's 256 columns

  v8f acc[16];
#pragma unroll
  for (int tt = 0; tt < 16; ++tt) acc[tt] = (v8f)(0.0f);

  for (int k0 = 0; k0 < DD; k0 += 32) {
    v16h a = load_frag(Kb, DD, m0 + r, k0, h);
#pragma unroll
    for (int tt = 0; tt < 16; ++tt) {
      v16h bq = load_frag(Qb, DD, t0 + tt * 16 + r, k0, h);
      acc[tt] = WMMA_F16(a, bq, acc[tt]);
    }
  }
#pragma unroll
  for (int tt = 0; tt < 16; ++tt) {
#pragma unroll
    for (int g = 0; g < 8; ++g)
      smem[(8 * h + g) * SS + t0 + tt * 16 + r] = acc[tt][g] * scale;
  }
  __syncthreads();

  // ---- row softmax: each wave handles rows wave and wave+8 ----
  for (int rr = wave; rr < 16; rr += 8) {
    float* row = smem + rr * SS;
    float mx = -3.402823466e+38f;
    for (int j = lane; j < SS; j += 32) mx = fmaxf(mx, row[j]);
#pragma unroll
    for (int o = 16; o >= 1; o >>= 1) mx = fmaxf(mx, __shfl_xor(mx, o, 32));
    float sum = 0.0f;
    for (int j = lane; j < SS; j += 32) {
      float e = __expf(row[j] - mx);
      row[j] = e;
      sum += e;
    }
#pragma unroll
    for (int o = 16; o >= 1; o >>= 1) sum += __shfl_xor(sum, o, 32);
    float inv = 1.0f / sum;
    half_t* prow = Ph + (size_t)b * SS * SS + (size_t)(m0 + rr) * SS;
    for (int j = lane; j < SS; j += 32) prow[j] = (half_t)(row[j] * inv);
  }
}

// ---------------------------------------------------------------------------
// 5) out[b] = P[b] @ V[b] : A = P [S,S] row-major f16, Bt = Vt [D,S] f16,
//    32x64 wave tiles, f32 output row-major [S,D].
// ---------------------------------------------------------------------------
__global__ __launch_bounds__(256) void attn_out(const half_t* __restrict__ Ph,
                                                const half_t* __restrict__ Vt,
                                                float* __restrict__ Out) {
  const int b = blockIdx.y;
  const half_t* A  = Ph + (size_t)b * SS * SS;
  const half_t* Bt = Vt + (size_t)b * DD * SS;
  float* O = Out + (size_t)b * SS * DD;

  const int lane = threadIdx.x & 31;
  const int wave = threadIdx.x >> 5;
  const int gw   = blockIdx.x * (blockDim.x >> 5) + wave;
  const int wavesN = DD >> 6;                 // 16
  const int mt = gw / wavesN;
  const int nt = gw % wavesN;
  const int m0 = mt * 32;
  const int n0 = nt * 64;
  if (m0 >= SS) return;
  const int r = lane & 15, h = lane >> 4;

  v8f acc[4][2];
#pragma unroll
  for (int j = 0; j < 4; ++j) {
    acc[j][0] = (v8f)(0.0f);
    acc[j][1] = (v8f)(0.0f);
  }

  for (int k0 = 0; k0 < SS; k0 += 32) {
    v16h a0 = load_frag(A,  SS, m0 + r,      k0, h);
    v16h a1 = load_frag(A,  SS, m0 + 16 + r, k0, h);
    v16h b0 = load_frag(Bt, SS, n0 + r,      k0, h);
    v16h b1 = load_frag(Bt, SS, n0 + 16 + r, k0, h);
    v16h b2 = load_frag(Bt, SS, n0 + 32 + r, k0, h);
    v16h b3 = load_frag(Bt, SS, n0 + 48 + r, k0, h);
    acc[0][0] = WMMA_F16(a0, b0, acc[0][0]);
    acc[0][1] = WMMA_F16(a1, b0, acc[0][1]);
    acc[1][0] = WMMA_F16(a0, b1, acc[1][0]);
    acc[1][1] = WMMA_F16(a1, b1, acc[1][1]);
    acc[2][0] = WMMA_F16(a0, b2, acc[2][0]);
    acc[2][1] = WMMA_F16(a1, b2, acc[2][1]);
    acc[3][0] = WMMA_F16(a0, b3, acc[3][0]);
    acc[3][1] = WMMA_F16(a1, b3, acc[3][1]);
  }

#pragma unroll
  for (int j = 0; j < 4; ++j) {
#pragma unroll
    for (int mi = 0; mi < 2; ++mi) {
#pragma unroll
      for (int g = 0; g < 8; ++g) {
        int m = m0 + 16 * mi + 8 * h + g;
        O[(size_t)m * DD + n0 + 16 * j + r] = acc[j][mi][g];
      }
    }
  }
}

// ---------------------------------------------------------------------------
// Host orchestration
// ---------------------------------------------------------------------------
extern "C" void kernel_launch(void* const* d_in, const int* in_sizes, int n_in,
                              void* d_out, int out_size, void* d_ws, size_t ws_size,
                              hipStream_t stream) {
  const float* x  = (const float*)d_in[0];
  const float* wk = (const float*)d_in[1];
  const float* wq = (const float*)d_in[2];
  const float* wv = (const float*)d_in[3];
  float* out = (float*)d_out;

  // workspace layout (bytes)
  char* ws = (char*)d_ws;
  const size_t xh_bytes = (size_t)BB * SS * DD * 2;      // 16 MB
  const size_t wt_bytes = (size_t)DD * DD * 2;           //  2 MB each
  half_t* xh  = (half_t*)(ws);
  half_t* wkt = (half_t*)(ws + xh_bytes);
  half_t* wqt = (half_t*)(ws + xh_bytes + wt_bytes);
  half_t* wvt = (half_t*)(ws + xh_bytes + 2 * wt_bytes);
  half_t* Kh  = (half_t*)(ws + xh_bytes + 3 * wt_bytes);
  half_t* Qh  = (half_t*)(ws + 2 * xh_bytes + 3 * wt_bytes);
  half_t* Vt  = (half_t*)(ws + 3 * xh_bytes + 3 * wt_bytes);
  half_t* Ph  = (half_t*)(ws + 4 * xh_bytes + 3 * wt_bytes);  // 32 MB

  // 1) x -> f16
  {
    int n4 = (BB * SS * DD) / 4;                       // 2,097,152
    cvt_f16<<<n4 / 256, 256, 0, stream>>>(x, xh, n4);
  }
  // 2) weights -> f16 transposed
  {
    dim3 g(DD / 32, DD / 32);
    transpose_w<<<g, 256, 0, stream>>>(wk, wkt);
    transpose_w<<<g, 256, 0, stream>>>(wq, wqt);
    transpose_w<<<g, 256, 0, stream>>>(wv, wvt);
  }
  // 3) projections: M = B*S = 8192, N = K = 1024, 32x64 wave tiles
  {
    const int M = BB * SS, N = DD, K = DD;
    int waves  = (M / 32) * (N / 64);                  // 4096
    int blocks = waves / 8;                            // 512
    gemm16<<<blocks, 256, 0, stream>>>(xh, wkt, Kh, M, N, K, 0);
    gemm16<<<blocks, 256, 0, stream>>>(xh, wqt, Qh, M, N, K, 0);
    gemm16<<<blocks, 256, 0, stream>>>(xh, wvt, Vt, M, N, K, 1);
  }
  // 4) scores + softmax (128 KB dynamic LDS per block)
  {
    dim3 g(SS / 16, BB);                               // (128, 4)
    size_t lds = (size_t)16 * SS * sizeof(float);      // 131072
    attn_scores<<<g, 256, lds, stream>>>(Kh, Qh, Ph, 0.03125f /* 1/sqrt(1024) */);
  }
  // 5) out = P @ V, 32x64 wave tiles
  {
    int waves  = (SS / 32) * (DD / 64);                // 1024 per batch
    dim3 g(waves / 8, BB);                             // (128, 4)
    attn_out<<<g, 256, 0, stream>>>(Ph, Vt, out);
  }
}
